// TorchAttractorLanguageModel_61168924230186
// MI455X (gfx1250) — matrix operand order repaired and checked by reading, not compiled, not run on hardware
//
#include <hip/hip_runtime.h>

// ---------------------------------------------------------------------------
// MI455X (gfx1250) attractor LM:
//   k1: signals for input tokens (LN + L2 norm)          -> ws.sigs  [T][B][D] f32
//   k2: diffusion f32 -> bf16 row-major [D][D]           -> ws.diffB
//   k3: full 960-step scan in ONE launch (16 blocks own 16 rows each in LDS),
//       per-step 16x512x512 GEMM via v_wmma_f32_16x16x32_bf16.
//       An asm pointer-barrier (on an explicit addrspace(1) pointer so the
//       loads stay GLOBAL_LOAD, LOADcnt-only) keeps diffB streaming from the
//       shared L2 image instead of being hoisted+spilled to scratch.
//   k4: logits = -cdist(state, signals(all V rows))/softplus(temp),
//       GEMM 256 x 50257 x 512 via bf16 WMMA; each block normalizes its 32
//       vocab rows exactly once (grid.y == 1) and each wave does 4 M-tiles
//       sharing one B operand.
// ---------------------------------------------------------------------------

typedef __bf16 bf16_t;
typedef __attribute__((ext_vector_type(16))) bf16_t v16bf;
typedef __attribute__((ext_vector_type(8)))  bf16_t v8bf;
typedef __attribute__((ext_vector_type(8)))  float  v8f;

typedef const bf16_t __attribute__((address_space(1)))* gbf16p;  // global ptr
typedef const v8bf   __attribute__((address_space(1)))* gv8p;

#define VOCAB   50257
#define DIM     512
#define BATCH   256
#define SEQ     32
#define NSTEP   30
#define DT_C    0.04f
#define CUBIC_C 0.008f
#define EPS_LN  1e-5f

__device__ __forceinline__ float wred(float v) {
#pragma unroll
  for (int off = 16; off > 0; off >>= 1) v += __shfl_xor(v, off, 32);
  return v;
}

// A operand: lane holds row l%16; 16 contiguous K halfwords.
__device__ __forceinline__ v16bf ld_a(const bf16_t* p) {
  v8bf lo = *(const v8bf*)p;
  v8bf hi = *(const v8bf*)(p + 8);
  return __builtin_shufflevector(lo, hi, 0,1,2,3,4,5,6,7,8,9,10,11,12,13,14,15);
}
// B operand: lane holds column l%16; 16 contiguous K halfwords.
__device__ __forceinline__ v16bf ld_b(const bf16_t* p) {
  v8bf lo = *(const v8bf*)p;
  v8bf hi = *(const v8bf*)(p + 8);
  return __builtin_shufflevector(lo, hi, 0,1,2,3,4,5,6,7,8,9,10,11,12,13,14,15);
}
// B operand from an explicit global-addrspace pointer -> global_load_b128.
__device__ __forceinline__ v16bf ld_b_g(gbf16p p) {
  v8bf lo = *(gv8p)p;
  v8bf hi = *(gv8p)(p + 8);
  return __builtin_shufflevector(lo, hi, 0,1,2,3,4,5,6,7,8,9,10,11,12,13,14,15);
}

// ------------------------- k1: token signals -------------------------------
__global__ void __launch_bounds__(256)
sig_kernel(const int* __restrict__ ids, const float* __restrict__ embed,
           float* __restrict__ sigs) {
  const int wave = threadIdx.x >> 5, lane = threadIdx.x & 31;
  const int row = blockIdx.x * 8 + wave;          // 0..8191
  const int b = row >> 5, t = row & 31;
  int id = ids[b * SEQ + t];
  id = max(0, min(VOCAB - 1, id));
  const float* src = embed + (size_t)id * DIM;
  float x[16], sum = 0.f, sq = 0.f;
#pragma unroll
  for (int j = 0; j < 16; ++j) { x[j] = src[j * 32 + lane]; sum += x[j]; sq += x[j] * x[j]; }
  sum = wred(sum); sq = wred(sq);
  const float mu   = sum * (1.f / 512.f);
  const float var  = fmaxf(sq * (1.f / 512.f) - mu * mu, 0.f);
  const float rstd = rsqrtf(var + EPS_LN);
  const float nrm  = sqrtf(512.f * var / (var + EPS_LN)); // ||LN(x)||
  const float scale = rstd / fmaxf(nrm, 1e-12f);
  float* dst = sigs + ((size_t)t * BATCH + b) * DIM;
#pragma unroll
  for (int j = 0; j < 16; ++j) dst[j * 32 + lane] = (x[j] - mu) * scale;
}

// ------------------------- k2: diffusion -> bf16 ---------------------------
__global__ void __launch_bounds__(256)
cvt_kernel(const float* __restrict__ src, bf16_t* __restrict__ dst) {
  const int idx = blockIdx.x * 256 + threadIdx.x;   // grid covers 512*512 exactly
  dst[idx] = (bf16_t)src[idx];
}

// ------------------------- k3: 960-step scan -------------------------------
__global__ void __launch_bounds__(256)
scan_kernel(const float* __restrict__ sigs, const bf16_t* __restrict__ diffB,
            bf16_t* __restrict__ stateB, float* __restrict__ snorm2) {
  __shared__ float  s[16][516];     // f32 state, padded rows
  __shared__ bf16_t sh[16][520];    // bf16 quantized state (WMMA A source)
  __shared__ float  mu[16];
  const int wave = threadIdx.x >> 5, lane = threadIdx.x & 31;
  const int b0 = blockIdx.x * 16;   // this block owns batch rows b0..b0+15
  const int n0 = wave * 64;         // this wave owns 64 output columns

  for (int idx = threadIdx.x; idx < 16 * 512; idx += 256)
    s[idx >> 9][idx & 511] = 0.f;
  __syncthreads();

  for (int t = 0; t < SEQ; ++t) {
    const float* sig = sigs + ((size_t)t * BATCH + b0) * DIM;
    for (int it = 0; it < NSTEP; ++it) {
      // phase 1: per-row mean + bf16 quantize (2 rows per wave)
#pragma unroll
      for (int rh = 0; rh < 2; ++rh) {
        const int rr = wave * 2 + rh;
        float x[16], sum = 0.f;
#pragma unroll
        for (int j = 0; j < 16; ++j) { x[j] = s[rr][j * 32 + lane]; sum += x[j]; }
        sum = wred(sum);
        if (lane == 0) mu[rr] = sum * (1.f / 512.f);
#pragma unroll
        for (int j = 0; j < 16; ++j) sh[rr][j * 32 + lane] = (bf16_t)x[j];
      }
      __syncthreads();

      // Opaque pointer barrier: stops the compiler from hoisting the (step-
      // invariant) diffB panel out of the 960-step loop and spilling 2KB/lane
      // to scratch. Explicit addrspace(1) keeps the loads GLOBAL (LOADcnt
      // only), decoupled from the DS counter used by the A-operand loads.
      gbf16p diffp = (gbf16p)diffB;
      asm volatile("" : "+s"(diffp));

      // phase 2: drift GEMM, 4 x (16x16) tiles per wave, K=512
      v8f acc[4];
#pragma unroll
      for (int nt = 0; nt < 4; ++nt) acc[nt] = (v8f){0.f,0.f,0.f,0.f,0.f,0.f,0.f,0.f};
      for (int kb = 0; kb < 512; kb += 32) {
        const v16bf a = ld_a(&sh[lane & 15][kb + ((lane >> 4) << 4)]);
#pragma unroll
        for (int nt = 0; nt < 4; ++nt) {
          gbf16p bp = diffp + (size_t)(n0 + nt * 16 + (lane & 15)) * DIM
                            + kb + ((lane >> 4) << 4);
          acc[nt] = __builtin_amdgcn_wmma_f32_16x16x32_bf16(
              false, a, false, ld_b_g(bp), (short)0, acc[nt], false, false);
        }
      }

      // phase 3: cubic + sig + clamped Euler update (own columns only)
#pragma unroll
      for (int nt = 0; nt < 4; ++nt) {
        const int n = n0 + nt * 16 + (lane & 15);
#pragma unroll
        for (int r = 0; r < 8; ++r) {
          const int m = r + ((lane >> 4) << 3);
          const float sold = s[m][n];
          const float c = sold - mu[m];
          const float dr = acc[nt][r] + CUBIC_C * c * c * c + sig[(size_t)m * DIM + n];
          float sn = sold + DT_C * dr;
          if (!__builtin_isfinite(sn)) sn = 0.f;            // nan_to_num(0,0,0)
          s[m][n] = fminf(fmaxf(sn, -80.f), 80.f);
        }
      }
      __syncthreads();
    }
  }

  // writeout: bf16 state + row norms^2 for the distance epilogue
#pragma unroll
  for (int rh = 0; rh < 2; ++rh) {
    const int rr = wave * 2 + rh;
    float sq = 0.f;
#pragma unroll
    for (int j = 0; j < 16; ++j) {
      const float v = s[rr][j * 32 + lane];
      sq += v * v;
      stateB[(size_t)(b0 + rr) * DIM + j * 32 + lane] = (bf16_t)v;
    }
    sq = wred(sq);
    if (lane == 0) snorm2[b0 + rr] = sq;
  }
}

// ------------------------- k4: logits --------------------------------------
// Grid: x = ceil(V/32) N-panels, y = 1 (all 256 state rows per block).
// Wave w: n-subtile = w>>2 (one of 2, 16 cols), m-subtiles (w&3)+4i, i=0..3.
__global__ void __launch_bounds__(256)
logits_kernel(const float* __restrict__ embed, const bf16_t* __restrict__ stateB,
              const float* __restrict__ snorm2, const float* __restrict__ traw,
              float* __restrict__ out) {
  __shared__ bf16_t sigsh[32][520];   // 32 normalized vocab rows (WMMA B source)
  __shared__ float  sq2[32];          // their squared norms (1 or degenerate)
  const int wave = threadIdx.x >> 5, lane = threadIdx.x & 31;
  const int n_base = blockIdx.x * 32;

  for (int rr = wave; rr < 32; rr += 8) {
    const int v = n_base + rr;
    if (v < VOCAB) {
      const float* src = embed + (size_t)v * DIM;
      float x[16], sum = 0.f, sq = 0.f;
#pragma unroll
      for (int j = 0; j < 16; ++j) { x[j] = src[j * 32 + lane]; sum += x[j]; sq += x[j] * x[j]; }
      sum = wred(sum); sq = wred(sq);
      const float mu   = sum * (1.f / 512.f);
      const float var  = fmaxf(sq * (1.f / 512.f) - mu * mu, 0.f);
      const float rstd = rsqrtf(var + EPS_LN);
      const float nrm  = sqrtf(512.f * var / (var + EPS_LN));
      const float invn = 1.f / fmaxf(nrm, 1e-12f);
      const float scale = rstd * invn;
#pragma unroll
      for (int j = 0; j < 16; ++j) sigsh[rr][j * 32 + lane] = (bf16_t)((x[j] - mu) * scale);
      if (lane == 0) { const float r = nrm * invn; sq2[rr] = r * r; }
    } else {
#pragma unroll
      for (int j = 0; j < 16; ++j) sigsh[rr][j * 32 + lane] = (bf16_t)0.f;
      if (lane == 0) sq2[rr] = 0.f;
    }
  }
  __syncthreads();

  const float sp = fmaxf(log1pf(expf(traw[0])), 1e-6f);
  const float invT = 1.f / sp;

  const int nsub = wave >> 2;          // 0..1
  const int mrow = wave & 3;           // m-subtiles mrow, mrow+4, mrow+8, mrow+12
  v8f acc[4];
#pragma unroll
  for (int i = 0; i < 4; ++i) acc[i] = (v8f){0.f,0.f,0.f,0.f,0.f,0.f,0.f,0.f};

  for (int kb = 0; kb < 512; kb += 32) {
    const bf16_t* bp = &sigsh[nsub * 16 + (lane & 15)][kb + ((lane >> 4) << 4)];
    const v16bf b = ld_b(bp);
#pragma unroll
    for (int i = 0; i < 4; ++i) {
      const int m0 = (mrow + 4 * i) * 16;
      const bf16_t* ap = stateB + (size_t)(m0 + (lane & 15)) * DIM + kb + ((lane >> 4) << 4);
      acc[i] = __builtin_amdgcn_wmma_f32_16x16x32_bf16(
          false, ld_a(ap), false, b, (short)0, acc[i], false, false);
    }
  }

#pragma unroll
  for (int i = 0; i < 4; ++i) {
    const int m0 = (mrow + 4 * i) * 16;
#pragma unroll
    for (int r = 0; r < 8; ++r) {
      const int m  = m0 + r + ((lane >> 4) << 3);
      const int nl = nsub * 16 + (lane & 15);
      const int n  = n_base + nl;
      const float d2 = snorm2[m] + sq2[nl] - 2.f * acc[i][r];
      const float dist = sqrtf(fmaxf(d2, 0.f));
      float lg = -dist * invT;
      if (!__builtin_isfinite(lg)) lg = __builtin_isnan(lg) ? 0.f : (lg > 0.f ? 0.f : -10000.f);
      if (n < VOCAB) out[(size_t)m * VOCAB + n] = lg;
    }
  }
}

// ------------------------- host launcher -----------------------------------
extern "C" void kernel_launch(void* const* d_in, const int* in_sizes, int n_in,
                              void* d_out, int out_size, void* d_ws, size_t ws_size,
                              hipStream_t stream) {
  (void)in_sizes; (void)n_in; (void)out_size; (void)ws_size;
  const int*   ids   = (const int*)d_in[0];
  const float* embed = (const float*)d_in[1];
  const float* diff  = (const float*)d_in[2];
  const float* traw  = (const float*)d_in[3];
  float* out = (float*)d_out;

  char* ws = (char*)d_ws;
  float*  sigs   = (float*)ws;                                   // 32*256*512 f32 = 16 MiB
  bf16_t* diffB  = (bf16_t*)(ws + (size_t)16777216);             // 512*512 bf16  = 512 KiB
  bf16_t* stateB = (bf16_t*)(ws + (size_t)16777216 + 524288);    // 256*512 bf16  = 256 KiB
  float*  snorm2 = (float*)(ws + (size_t)16777216 + 524288 + 262144); // 256 f32

  sig_kernel<<<1024, 256, 0, stream>>>(ids, embed, sigs);
  cvt_kernel<<<1024, 256, 0, stream>>>(diff, diffB);
  scan_kernel<<<16, 256, 0, stream>>>(sigs, diffB, stateB, snorm2);
  dim3 g4((VOCAB + 31) / 32, 1);
  logits_kernel<<<g4, 256, 0, stream>>>(embed, stateB, snorm2, traw, out);
}
